// SelfAttv1_51634096833128
// MI455X (gfx1250) — compile-verified, matrix operand order checked by reading, not run on hardware
//
#include <hip/hip_runtime.h>
#include <hip/hip_bf16.h>

// ---------------------------------------------------------------------------
// Self-attention for MI455X (gfx1250, wave32, WMMA).
// B=4, S=4096, H=1024.  All matmuls run on v_wmma_f32_16x16x32_bf16.
// Projection GEMM stages W tiles with GLOBAL_LOAD_ASYNC_TO_LDS (ASYNCcnt)
// and double-buffers LDS so the copy overlaps the WMMAs.
// Attention uses 64-row query tiles (1024 threads = 32 waves: 4 query
// groups x 8 hidden-dim slices) to raise arithmetic intensity on K/V.
// ---------------------------------------------------------------------------

typedef __attribute__((ext_vector_type(16))) __bf16 v16bf;
typedef __attribute__((ext_vector_type(8)))  float  v8f;
typedef __attribute__((ext_vector_type(8)))  __bf16 v8bf;

#define BATCH 4
#define SEQ   4096
#define HID   1024
#define MROWS (BATCH * SEQ)   // 16384
#define QTILE 64              // query rows per attention block

#if defined(__AMDGCN__) && __has_builtin(__builtin_amdgcn_global_load_async_to_lds_b128)
#define HAVE_ASYNC_LDS 1
#else
#define HAVE_ASYNC_LDS 0
#endif

#if HAVE_ASYNC_LDS
// builtin signature (from hipcc diagnostic): arg0 = int4* in AS1 (global),
// arg1 = int4* in AS3 (LDS), then two integer immediates (offset, cpol).
typedef __attribute__((__vector_size__(16))) int i32x4;
typedef __attribute__((address_space(1))) i32x4 i32x4_g;
typedef __attribute__((address_space(3))) i32x4 i32x4_s;
#endif

static __device__ __forceinline__ v8f wmma_bf16(v16bf a, v16bf b, v8f c) {
    return __builtin_amdgcn_wmma_f32_16x16x32_bf16(
        /*neg_a=*/false, a, /*neg_b=*/false, b,
        /*c_mod=*/(short)0, c, /*reuse_a=*/false, /*reuse_b=*/false);
}

// copy 16 bf16 (32B) from global to LDS, async if the CDNA5 path is available
static __device__ __forceinline__ void async_copy32B(const __bf16* g, __bf16* l) {
#if HAVE_ASYNC_LDS
    i32x4_g* gp = (i32x4_g*)(void*)g;
    i32x4_s* lp = (i32x4_s*)(void*)l;
    __builtin_amdgcn_global_load_async_to_lds_b128(gp,     lp,     0, 0);
    __builtin_amdgcn_global_load_async_to_lds_b128(gp + 1, lp + 1, 0, 0);
#else
    *(uint4*)l       = *(const uint4*)g;
    *(uint4*)(l + 8) = *(const uint4*)(g + 8);
#endif
}

static __device__ __forceinline__ void wait_async_copies() {
#if HAVE_ASYNC_LDS
#if __has_builtin(__builtin_amdgcn_s_wait_asynccnt)
    __builtin_amdgcn_s_wait_asynccnt(0);
#endif
#endif
}

// ---------------------------------------------------------------------------
// f32 -> bf16 conversion, 8 elements / thread, b128 loads.
// ---------------------------------------------------------------------------
__global__ void __launch_bounds__(256) cvt_bf16_kernel(const float* __restrict__ in,
                                                       unsigned short* __restrict__ out_u16,
                                                       int n) {
    int i = (blockIdx.x * blockDim.x + threadIdx.x) * 8;
    if (i >= n) return;
    float4 a = *(const float4*)(in + i);
    float4 b = *(const float4*)(in + i + 4);
    v8bf o;
    o[0] = (__bf16)a.x; o[1] = (__bf16)a.y; o[2] = (__bf16)a.z; o[3] = (__bf16)a.w;
    o[4] = (__bf16)b.x; o[5] = (__bf16)b.y; o[6] = (__bf16)b.z; o[7] = (__bf16)b.w;
    *(v8bf*)((__bf16*)out_u16 + i) = o;
}

// ---------------------------------------------------------------------------
// Projection: Y[m,n] = sum_k Xb[m,k] * Wb[n,k] + bias[n]   (nn.Linear, x@W^T+b)
// Block: 256 threads (8 waves). Tile: M=128 (16 rows/wave), N=128.
// W tile (128x32 bf16 = 8KB) double-buffered in LDS via async copies.
// transposed==0: Y stored [m][n] bf16 (Q,K).
// transposed==1: Y stored as Vt[b][n][s] bf16 (V, pre-transposed for PV GEMM).
// ---------------------------------------------------------------------------
__global__ void __launch_bounds__(256, 2)
proj_kernel(const unsigned short* __restrict__ Xb_u16,
            const unsigned short* __restrict__ Wb_u16,
            const float* __restrict__ bias,
            unsigned short* __restrict__ out_u16,
            int transposed) {
    const __bf16* Xb = (const __bf16*)Xb_u16;
    const __bf16* Wb = (const __bf16*)Wb_u16;
    __bf16*       Yb = (__bf16*)out_u16;

    __shared__ __align__(32) __bf16 wt[2][128][32];   // double-buffered [n][k] W tile

    const int lane = threadIdx.x & 31;
    const int wave = threadIdx.x >> 5;
    const int lm   = lane & 15;
    const int lh   = lane >> 4;
    const int khalf = lh * 16;

    const int m0 = blockIdx.x * 128;    // gridDim.x = 128
    const int n0 = blockIdx.y * 128;    // gridDim.y = 8
    const int mrow = m0 + wave * 16 + lm;

    // staging assignment: each thread copies 32B (one half-row of the tile)
    const int st_row = threadIdx.x >> 1;          // 0..127
    const int st_col = (threadIdx.x & 1) * 16;    // 0 or 16
    const __bf16* wsrc = Wb + (size_t)(n0 + st_row) * HID + st_col;

    v8f acc[8] = {};

    // prologue: stage k-step 0 into buffer 0
    async_copy32B(wsrc, &wt[0][st_row][st_col]);

    for (int kk = 0; kk < HID / 32; ++kk) {
        wait_async_copies();
        __syncthreads();                 // buffer kk&1 ready for everyone

        // stage next k-step into the other buffer (overlaps the WMMAs below)
        if (kk + 1 < HID / 32)
            async_copy32B(wsrc + (kk + 1) * 32, &wt[(kk + 1) & 1][st_row][st_col]);

        const int buf = kk & 1;
        v16bf a = *(const v16bf*)&Xb[(size_t)mrow * HID + kk * 32 + khalf];
#pragma unroll
        for (int nt = 0; nt < 8; ++nt) {
            v16bf b = *(const v16bf*)&wt[buf][nt * 16 + lm][khalf];
            acc[nt] = wmma_bf16(a, b, acc[nt]);
        }
        // no trailing barrier needed: every wave's LDS reads are consumed by a
        // WMMA before it reaches the next top barrier, and the buffer being
        // overwritten next was last read a full barrier ago.
    }

    // --- epilogue: bias add + store ---
#pragma unroll
    for (int nt = 0; nt < 8; ++nt) {
        int n = n0 + nt * 16 + lm;
        float bv = bias[n];
#pragma unroll
        for (int r = 0; r < 8; ++r) {
            int row = m0 + wave * 16 + r + lh * 8;   // C layout: M = r + 8*(lane/16)
            float v = acc[nt][r] + bv;
            if (!transposed) {
                Yb[(size_t)row * HID + n] = (__bf16)v;
            } else {
                int b = row >> 12;            // row / 4096
                int s = row & (SEQ - 1);
                Yb[((size_t)b * HID + n) * SEQ + s] = (__bf16)v;
            }
        }
    }
}

// ---------------------------------------------------------------------------
// Flash attention: 64-row query tile per block, 1024 threads = 32 waves.
// Wave w: query group qg = w>>3 (16 rows), hidden slice dsl = w&7 (128 dims).
// Key tiles of 32; online softmax for the 64 rows done by threads 0..63.
// ---------------------------------------------------------------------------
__global__ void __launch_bounds__(1024, 1)
attn_kernel(const unsigned short* __restrict__ Qb_u16,
            const unsigned short* __restrict__ Kb_u16,
            const unsigned short* __restrict__ Vt_u16,
            float* __restrict__ out) {
    const __bf16* Qb = (const __bf16*)Qb_u16;
    const __bf16* Kb = (const __bf16*)Kb_u16;
    const __bf16* Vt = (const __bf16*)Vt_u16;

    __shared__ __align__(32) float  sc[QTILE][32];   // cross-wave score reduction
    __shared__ __align__(32) __bf16 P[QTILE][32];    // softmax probs (bf16)
    __shared__ float m_s[QTILE], l_s[QTILE], esc[QTILE];

    const int lane = threadIdx.x & 31;
    const int wave = threadIdx.x >> 5;
    const int lm   = lane & 15;
    const int lh   = lane >> 4;
    const int khalf = lh * 16;

    const int qg    = wave >> 3;         // query group 0..3 (16 rows each)
    const int dsl   = wave & 7;          // hidden-dim slice 0..7
    const int dbase = dsl * 128;

    const int qt    = blockIdx.x;                  // 256 tiles of 64 rows
    const int batch = qt >> 6;                     // 64 tiles per batch
    const int s0    = (qt & 63) * QTILE;
    const int qrow0 = s0 + qg * 16;                // this wave's 16 query rows

    // init LDS state
    if (threadIdx.x < QTILE) { m_s[threadIdx.x] = -1e30f; l_s[threadIdx.x] = 0.0f; }
    ((float*)sc)[threadIdx.x]         = 0.0f;
    ((float*)sc)[threadIdx.x + 1024]  = 0.0f;

    // preload Q fragments: 16 rows x 128 d-slice (4 K-steps)
    v16bf qf[4];
    {
        const __bf16* qrow = Qb + ((size_t)(batch * SEQ + qrow0 + lm)) * HID + dbase + khalf;
#pragma unroll
        for (int kk = 0; kk < 4; ++kk) qf[kk] = *(const v16bf*)(qrow + kk * 32);
    }

    v8f oacc[8] = {};
    __syncthreads();

    const float sm_scale = 0.03125f;     // 1/sqrt(1024)

    for (int t0 = 0; t0 < SEQ; t0 += 32) {
        // prefetch next key tile (global_prefetch_b8)
        if (t0 + 32 < SEQ)
            __builtin_prefetch(Kb + ((size_t)(batch * SEQ + t0 + 32 + lm)) * HID + dbase, 0, 3);

        // ---- partial scores over this wave's 128-wide d-slice ----
        v8f cs[2] = {};
#pragma unroll
        for (int kk = 0; kk < 4; ++kk) {
#pragma unroll
            for (int nt = 0; nt < 2; ++nt) {
                v16bf kf = *(const v16bf*)(Kb +
                    ((size_t)(batch * SEQ + t0 + nt * 16 + lm)) * HID + dbase + kk * 32 + khalf);
                cs[nt] = wmma_bf16(qf[kk], kf, cs[nt]);
            }
        }
        // reduce partials across the 8 d-slice waves of this query group
#pragma unroll
        for (int nt = 0; nt < 2; ++nt)
#pragma unroll
            for (int r = 0; r < 8; ++r)
                atomicAdd(&sc[qg * 16 + r + lh * 8][nt * 16 + lm], cs[nt][r]);
        __syncthreads();

        // ---- online softmax: one lane per row, threads 0..63 (2 waves) ----
        if (threadIdx.x < QTILE) {
            int row = threadIdx.x;
            float mold = m_s[row];
            float mx = mold;
            float v[32];
#pragma unroll
            for (int t = 0; t < 32; ++t) {
                v[t] = sc[row][t] * sm_scale;
                mx = fmaxf(mx, v[t]);
            }
            float scale = __expf(mold - mx);
            float sum = 0.0f;
#pragma unroll
            for (int t = 0; t < 32; ++t) {
                float e = __expf(v[t] - mx);
                P[row][t] = (__bf16)e;
                sum += e;
                sc[row][t] = 0.0f;       // re-zero for next key tile
            }
            m_s[row] = mx;
            l_s[row] = l_s[row] * scale + sum;
            esc[row] = scale;
        }
        __syncthreads();

        // ---- rescale accumulators, then PV WMMA over this wave's slice ----
#pragma unroll
        for (int r = 0; r < 8; ++r) {
            float s = esc[qg * 16 + r + lh * 8];
#pragma unroll
            for (int nt = 0; nt < 8; ++nt) oacc[nt][r] *= s;
        }
        v16bf pf = *(const v16bf*)&P[qg * 16 + lm][khalf];   // A fragment of probs
#pragma unroll
        for (int nt = 0; nt < 8; ++nt) {
            v16bf vf = *(const v16bf*)(Vt +
                ((size_t)batch * HID + dbase + nt * 16 + lm) * SEQ + t0 + khalf);
            oacc[nt] = wmma_bf16(pf, vf, oacc[nt]);
        }
        __syncthreads();
    }

    // ---- final normalization + f32 store ----
#pragma unroll
    for (int r = 0; r < 8; ++r) {
        float linv = 1.0f / l_s[qg * 16 + r + lh * 8];
        int row = batch * SEQ + s0 + qg * 16 + r + lh * 8;
#pragma unroll
        for (int nt = 0; nt < 8; ++nt)
            out[(size_t)row * HID + dbase + nt * 16 + lm] = oacc[nt][r] * linv;
    }
}

// ---------------------------------------------------------------------------
// Host launcher
// ---------------------------------------------------------------------------
extern "C" void kernel_launch(void* const* d_in, const int* in_sizes, int n_in,
                              void* d_out, int out_size, void* d_ws, size_t ws_size,
                              hipStream_t stream) {
    const float* X  = (const float*)d_in[0];
    const float* Wq = (const float*)d_in[1];
    const float* bq = (const float*)d_in[2];
    const float* Wk = (const float*)d_in[3];
    const float* bk = (const float*)d_in[4];
    const float* Wv = (const float*)d_in[5];
    const float* bv = (const float*)d_in[6];
    float* out = (float*)d_out;

    // workspace layout (bf16 buffers), all sizes in bytes
    char* ws = (char*)d_ws;
    const size_t XB_BYTES = (size_t)MROWS * HID * 2;          // 32 MB
    const size_t W_BYTES  = (size_t)HID * HID * 2;            // 2 MB
    const size_t QB_BYTES = XB_BYTES;                         // 32 MB each
    unsigned short* Xb  = (unsigned short*)(ws);
    unsigned short* Wqb = (unsigned short*)(ws + XB_BYTES);
    unsigned short* Wkb = (unsigned short*)(ws + XB_BYTES + W_BYTES);
    unsigned short* Wvb = (unsigned short*)(ws + XB_BYTES + 2 * W_BYTES);
    unsigned short* Qb  = (unsigned short*)(ws + XB_BYTES + 3 * W_BYTES);
    unsigned short* Kb  = (unsigned short*)(ws + XB_BYTES + 3 * W_BYTES + QB_BYTES);
    unsigned short* Vtb = (unsigned short*)(ws + XB_BYTES + 3 * W_BYTES + 2 * QB_BYTES);

    // 1) f32 -> bf16 conversions
    {
        int nX = MROWS * HID;                    // 16,777,216
        int nW = HID * HID;                      // 1,048,576
        cvt_bf16_kernel<<<nX / (256 * 8), 256, 0, stream>>>(X,  Xb,  nX);
        cvt_bf16_kernel<<<nW / (256 * 8), 256, 0, stream>>>(Wq, Wqb, nW);
        cvt_bf16_kernel<<<nW / (256 * 8), 256, 0, stream>>>(Wk, Wkb, nW);
        cvt_bf16_kernel<<<nW / (256 * 8), 256, 0, stream>>>(Wv, Wvb, nW);
    }

    // 2) QKV projections (V stored transposed for the PV GEMM)
    {
        dim3 grid(MROWS / 128, HID / 128);       // 128 x 8
        proj_kernel<<<grid, 256, 0, stream>>>(Xb, Wqb, bq, Qb,  0);
        proj_kernel<<<grid, 256, 0, stream>>>(Xb, Wkb, bk, Kb,  0);
        proj_kernel<<<grid, 256, 0, stream>>>(Xb, Wvb, bv, Vtb, 1);
    }

    // 3) flash attention (64-row query tiles, 1024 threads)
    attn_kernel<<<MROWS / QTILE, 1024, 0, stream>>>(Qb, Kb, Vtb, out);
}
